// MDescAug_25718264169168
// MI455X (gfx1250) — matrix operand order repaired and checked by reading, not compiled, not run on hardware
//
#include <hip/hip_runtime.h>
#include <hip/hip_bf16.h>

typedef __attribute__((ext_vector_type(16))) __bf16 v16bf;
typedef __attribute__((ext_vector_type(8)))  __bf16 v8bf;
typedef __attribute__((ext_vector_type(8)))  float  v8f;

#define NQ    64
#define MM    400
#define TOPK  10
#define DIM   2048
#define NROWS 10000
#define BETA  0.15f
#define NEG_INF (-3.0e38f)

// ---------------------------------------------------------------------------
// Kernel 1: fp32 -> bf16 (round to nearest even), X -> Xb in workspace.
// ---------------------------------------------------------------------------
__device__ __forceinline__ unsigned short f2bf_rne(float f) {
    unsigned u = __float_as_uint(f);
    u += 0x7FFFu + ((u >> 16) & 1u);
    return (unsigned short)(u >> 16);
}

__global__ __launch_bounds__(256) void convert_bf16(
    const float* __restrict__ X, unsigned short* __restrict__ Xb, int n)
{
    int i = (blockIdx.x * 256 + threadIdx.x) * 8;
    if (i >= n) return;
    const float4* p = (const float4*)(X + i);
    float4 a = p[0], b = p[1];
    uint4 o;
    o.x = (unsigned)f2bf_rne(a.x) | ((unsigned)f2bf_rne(a.y) << 16);
    o.y = (unsigned)f2bf_rne(a.z) | ((unsigned)f2bf_rne(a.w) << 16);
    o.z = (unsigned)f2bf_rne(b.x) | ((unsigned)f2bf_rne(b.y) << 16);
    o.w = (unsigned)f2bf_rne(b.z) | ((unsigned)f2bf_rne(b.w) << 16);
    *(uint4*)(Xb + i) = o;
}

// ---------------------------------------------------------------------------
// Kernel 2: per (query a, 16-row tile) compute sim rows = G G^T with
// v_wmma_f32_16x16x32_bf16, stage the 16x400 tile in LDS, then top-10/row.
// 128 threads = 4 waves; wave w owns N-tiles w, w+4, w+8, ...
// ---------------------------------------------------------------------------
__global__ __launch_bounds__(128) void sim_topk(
    const __bf16* __restrict__ Xb, const long long* __restrict__ ranks,
    float* __restrict__ vals, int* __restrict__ idx)
{
    __shared__ float    sim[16][MM];
    __shared__ unsigned rowidx[MM];

    const int a  = blockIdx.x / 25;
    const int i0 = (blockIdx.x % 25) * 16;
    const int tid = threadIdx.x;

    for (int r = tid; r < MM; r += 128)
        rowidx[r] = (unsigned)ranks[(long long)r * NQ + a];   // rt[a][r]
    __syncthreads();

    const int lane = tid & 31;
    const int wave = tid >> 5;
    const int half = lane >> 4;      // 0: lanes 0-15, 1: lanes 16-31
    const int mn   = lane & 15;      // M for A-frag, N for B-frag

    // A fragment rows are fixed for the whole block's tile.
    const __bf16* Arow = Xb + (size_t)rowidx[i0 + mn] * DIM;

    for (int jt = wave; jt < 25; jt += 4) {
        const int j0 = jt * 16;
        const __bf16* Brow = Xb + (size_t)rowidx[j0 + mn] * DIM;

        v8f acc = {};
        #pragma unroll 2
        for (int k = 0; k < DIM; k += 32) {
            // A 16x32: lane holds row M=mn, K = half*16 + e (contiguous, matches B)
            v8bf alo = *(const v8bf*)(Arow + k + half * 16);
            v8bf ahi = *(const v8bf*)(Arow + k + half * 16 + 8);
            v16bf afrag = __builtin_shufflevector(alo, ahi,
                0,1,2,3,4,5,6,7,8,9,10,11,12,13,14,15);
            // B 32x16: lane holds col N=mn, K = half*16 + e (contiguous 32B)
            v16bf bfrag = *(const v16bf*)(Brow + k + half * 16);
            acc = __builtin_amdgcn_wmma_f32_16x16x32_bf16(
                false, afrag, false, bfrag, (short)0, acc, false, false);
        }
        // C/D layout: VGPR r, lane l -> M = r + 8*(l/16), N = l%16
        #pragma unroll
        for (int r = 0; r < 8; ++r)
            sim[r + half * 8][j0 + mn] = acc[r];
    }
    __syncthreads();

    // Top-10 per row; strict '>' keeps the earliest index on ties (top_k tie rule).
    if (tid < 16) {
        const size_t base = ((size_t)a * MM + (i0 + tid)) * TOPK;
        for (int t = 0; t < TOPK; ++t) {
            float best = NEG_INF; int bi = 0;
            for (int j = 0; j < MM; ++j) {
                float v = sim[tid][j];
                if (v > best) { best = v; bi = j; }
            }
            sim[tid][bi] = NEG_INF;
            vals[base + t] = best;
            idx [base + t] = bi;
        }
    }
}

// ---------------------------------------------------------------------------
// Kernel 3: x_dba[a,b,:] = (sum_t w_t * X[rt[a,idx_t]]) / sum_t w_t  (fp32),
// fused with scores[a,b] = Q[a] . x_dba[a,b].  One block per (a,b).
// ---------------------------------------------------------------------------
__global__ __launch_bounds__(256) void xdba_scores(
    const float* __restrict__ X, const float* __restrict__ Q,
    const long long* __restrict__ ranks,
    const float* __restrict__ vals, const int* __restrict__ idx,
    float* __restrict__ out_scores, float* __restrict__ out_xdba)
{
    const int ab = blockIdx.x;        // a*400 + b
    const int a  = ab / MM;
    const int tid = threadIdx.x;

    __shared__ float    wsh[TOPK];
    __shared__ unsigned grow[TOPK];
    __shared__ float    red[256];

    if (tid < TOPK) {
        float v = vals[(size_t)ab * TOPK + tid];
        wsh[tid] = (tid == 0) ? 1.0f : BETA * v;          // w0=1, wk=beta*valk
        int j = idx[(size_t)ab * TOPK + tid];
        grow[tid] = (unsigned)ranks[(long long)j * NQ + a]; // rt[a][idx_t]
    }
    __syncthreads();

    float wsum = 0.0f;
    #pragma unroll
    for (int t = 0; t < TOPK; ++t) wsum += wsh[t];
    const float inv = 1.0f / wsum;

    const float4* q4 = (const float4*)(Q + (size_t)a * DIM);
    float4* o4 = (float4*)(out_xdba + (size_t)ab * DIM);
    float partial = 0.0f;

    #pragma unroll
    for (int c = 0; c < 2; ++c) {                  // 512 float4 / 256 threads
        const int d4 = tid + c * 256;
        float4 accv = {0.f, 0.f, 0.f, 0.f};
        #pragma unroll
        for (int t = 0; t < TOPK; ++t) {
            float4 xv = ((const float4*)(X + (size_t)grow[t] * DIM))[d4];
            float w = wsh[t];
            accv.x += w * xv.x; accv.y += w * xv.y;
            accv.z += w * xv.z; accv.w += w * xv.w;
        }
        accv.x *= inv; accv.y *= inv; accv.z *= inv; accv.w *= inv;
        o4[d4] = accv;
        float4 qv = q4[d4];
        partial += accv.x * qv.x + accv.y * qv.y + accv.z * qv.z + accv.w * qv.w;
    }

    red[tid] = partial;
    __syncthreads();
    for (int s = 128; s > 0; s >>= 1) {
        if (tid < s) red[tid] += red[tid + s];
        __syncthreads();
    }
    if (tid == 0) out_scores[ab] = red[0];
}

// ---------------------------------------------------------------------------
// Kernel 4: stable descending rank-by-counting == argsort(-scores), then
// pre[a,rank]=b and final[a,rank]=rt[a,b].  One block per query.
// ---------------------------------------------------------------------------
__global__ __launch_bounds__(512) void rank_sort(
    const float* __restrict__ scores, const long long* __restrict__ ranks,
    float* __restrict__ out_final, float* __restrict__ out_pre)
{
    const int a = blockIdx.x;
    const int tid = threadIdx.x;
    __shared__ float s[MM];
    for (int r = tid; r < MM; r += 512) s[r] = scores[(size_t)a * MM + r];
    __syncthreads();
    if (tid < MM) {
        const float mine = s[tid];
        int rank = 0;
        for (int j = 0; j < MM; ++j) {
            float v = s[j];
            rank += (v > mine) || ((v == mine) && (j < tid));
        }
        out_pre  [(size_t)a * MM + rank] = (float)tid;
        out_final[(size_t)a * MM + rank] = (float)ranks[(long long)tid * NQ + a];
    }
}

// ---------------------------------------------------------------------------
extern "C" void kernel_launch(void* const* d_in, const int* in_sizes, int n_in,
                              void* d_out, int out_size, void* d_ws, size_t ws_size,
                              hipStream_t stream)
{
    const float*     X     = (const float*)d_in[0];       // (10000, 2048) f32
    const float*     Q     = (const float*)d_in[1];       // (64, 2048)    f32
    const long long* ranks = (const long long*)d_in[2];   // (10000, 64)   i64

    float* out        = (float*)d_out;                    // tuple flattened
    float* out_final  = out;                              // 25600
    float* out_scores = out + NQ * MM;                    // 25600
    float* out_pre    = out + 2 * NQ * MM;                // 25600
    float* out_xdba   = out + 3 * NQ * MM;                // 52428800

    // Workspace layout (all 256B aligned): Xbf16 | vals | idx
    char* ws = (char*)d_ws;
    unsigned short* Xb_u = (unsigned short*)ws;                       // 40,960,000 B
    float* vals = (float*)(ws + 40960000);                            //  1,024,000 B
    int*   idxb = (int*)  (ws + 40960000 + 1024000);                  //  1,024,000 B

    const int n = NROWS * DIM;                                        // 20,480,000
    convert_bf16<<<n / (256 * 8), 256, 0, stream>>>(X, Xb_u, n);
    sim_topk    <<<NQ * 25, 128, 0, stream>>>((const __bf16*)Xb_u, ranks, vals, idxb);
    xdba_scores <<<NQ * MM, 256, 0, stream>>>(X, Q, ranks, vals, idxb,
                                              out_scores, out_xdba);
    rank_sort   <<<NQ, 512, 0, stream>>>(out_scores, ranks, out_final, out_pre);
}
